// GCN_25795573580231
// MI455X (gfx1250) — matrix operand order repaired and checked by reading, not compiled, not run on hardware
//
#include <hip/hip_runtime.h>

// ---------------------------------------------------------------------------
// Types for CDNA5 WMMA (wave32): V_WMMA_F32_16X16X32_BF16
// ---------------------------------------------------------------------------
typedef __attribute__((ext_vector_type(16))) __bf16         v16bf;
typedef __attribute__((ext_vector_type(8)))  float          v8f;
typedef __attribute__((ext_vector_type(4)))  float          v4f;
typedef __attribute__((ext_vector_type(16))) unsigned short v16us;
typedef __attribute__((ext_vector_type(8)))  unsigned short v8us;
typedef __attribute__((ext_vector_type(4)))  unsigned short v4us;

__device__ __forceinline__ unsigned short f32_to_bf16_rne(float f) {
    unsigned int u = __builtin_bit_cast(unsigned int, f);
    u += 0x7FFFu + ((u >> 16) & 1u);   // round-to-nearest-even
    return (unsigned short)(u >> 16);
}

// ---------------------------------------------------------------------------
// 1) Weight transpose + fp32 -> bf16:  Wt[n*K + k] = bf16(W[k*N + n])
//    Gives contiguous-per-lane B fragments (2 x b128 loads) in the GEMM.
// ---------------------------------------------------------------------------
__global__ void transpose_bf16_kernel(const float* __restrict__ W,
                                      unsigned short* __restrict__ Wt,
                                      int rows /*K*/, int cols /*N*/) {
    int idx = blockIdx.x * blockDim.x + threadIdx.x;
    if (idx < rows * cols) {
        int k = idx / cols;
        int n = idx - k * cols;
        Wt[(size_t)n * rows + k] = f32_to_bf16_rne(W[idx]);
    }
}

// ---------------------------------------------------------------------------
// 2) CSR-by-destination build: zero -> histogram -> scan -> scatter
// ---------------------------------------------------------------------------
__global__ void zero_int_kernel(int* __restrict__ p, int n) {
    int i = blockIdx.x * blockDim.x + threadIdx.x;
    if (i < n) p[i] = 0;
}

__global__ void hist_kernel(const int* __restrict__ edge_dst,
                            int* __restrict__ deg, int e) {
    int i = blockIdx.x * blockDim.x + threadIdx.x;
    if (i < e) atomicAdd(&deg[edge_dst[i]], 1);
}

__global__ __launch_bounds__(1024)
void scan_kernel(const int* __restrict__ deg, int* __restrict__ off,
                 int* __restrict__ cursor, int n) {
    __shared__ int sdata[1024];
    __shared__ int carry;
    if (threadIdx.x == 0) carry = 0;
    __syncthreads();
    for (int base = 0; base < n; base += 1024) {
        int i = base + (int)threadIdx.x;
        int v = (i < n) ? deg[i] : 0;
        sdata[threadIdx.x] = v;
        __syncthreads();
        for (int ofs = 1; ofs < 1024; ofs <<= 1) {
            int t = (threadIdx.x >= (unsigned)ofs) ? sdata[threadIdx.x - ofs] : 0;
            __syncthreads();
            sdata[threadIdx.x] += t;
            __syncthreads();
        }
        int incl = sdata[threadIdx.x];
        int excl = carry + incl - v;
        if (i < n) { off[i] = excl; cursor[i] = excl; }
        int total = sdata[1023];
        __syncthreads();
        if (threadIdx.x == 0) carry += total;
        __syncthreads();
    }
    if (threadIdx.x == 0) off[n] = carry;
}

__global__ void scatter_kernel(const int* __restrict__ edge_dst,
                               int* __restrict__ cursor,
                               int* __restrict__ eidx, int e) {
    int i = blockIdx.x * blockDim.x + threadIdx.x;
    if (i < e) {
        int d = edge_dst[i];
        int pos = atomicAdd(&cursor[d], 1);
        eidx[pos] = i;
    }
}

// ---------------------------------------------------------------------------
// 3) Tiled WMMA GEMM:  C[M x NCOLS] = A[M x K] * B[K x NCOLS]
//    - 4 waves / block; block owns a 64-row M slab staged in LDS as bf16.
//    - B given pre-transposed Bt[NCOLS][K]: per-lane contiguous fragments.
//    - wave w computes 4 M-tiles x (NT/4) N-tiles -> W read once per BLOCK.
//    Fragment layouts used here (wave32, contiguous k for A and B):
//      A: lane l -> row l%16; K = 16*(l/16)+{0..15}
//      B: lane l -> col l%16; K = 16*(l/16)+{0..15} (contiguous in Bt)
//      C: lane l -> col l%16; vgpr i -> row 8*(l/16)+i
// ---------------------------------------------------------------------------
template <int K, int NCOLS, bool AF32>
__global__ __launch_bounds__(128)
void wmma_gemm_tiled(const void* __restrict__ Aany,
                     const unsigned short* __restrict__ Bt,
                     float* __restrict__ C, int M) {
    constexpr int KT  = K / 32;
    constexpr int NT  = NCOLS / 16;
    constexpr int NTW = NT / 4;            // N-tiles per wave
    constexpr int LDA = K + 8;             // padded LDS stride (bank spread)
    __shared__ unsigned short As[64 * LDA];

    const int tid  = threadIdx.x;          // 0..127
    const int lane = tid & 31;
    const int wave = tid >> 5;
    const int m00  = blockIdx.x * 64;

    // ---- stage A slab (64 x K) into LDS as bf16 ----
    if constexpr (AF32) {
        const float* Af = (const float*)Aany;
        for (int idx = tid * 4; idx < 64 * K; idx += 128 * 4) {
            int row = idx / K;
            int col = idx - row * K;
            v4f xv = {0.f, 0.f, 0.f, 0.f};
            if (m00 + row < M)
                xv = *(const v4f*)(Af + (size_t)(m00 + row) * K + col);
            v4us p;
            p[0] = f32_to_bf16_rne(xv[0]);
            p[1] = f32_to_bf16_rne(xv[1]);
            p[2] = f32_to_bf16_rne(xv[2]);
            p[3] = f32_to_bf16_rne(xv[3]);
            *(v4us*)&As[row * LDA + col] = p;
        }
    } else {
        const unsigned short* Ab = (const unsigned short*)Aany;
        for (int idx = tid * 8; idx < 64 * K; idx += 128 * 8) {
            int row = idx / K;
            int col = idx - row * K;
            v8us v = {0, 0, 0, 0, 0, 0, 0, 0};
            if (m00 + row < M)
                v = *(const v8us*)(Ab + (size_t)(m00 + row) * K + col);
            *(v8us*)&As[row * LDA + col] = v;
        }
    }
    __syncthreads();

    const int arow = lane & 15;
    const int akh  = (lane >> 4) * 16;
    const int bcol = lane & 15;
    const int bkh  = (lane >> 4) * 16;

    const v8f vzero = {0.f, 0.f, 0.f, 0.f, 0.f, 0.f, 0.f, 0.f};
    v8f acc[4][NTW];
#pragma unroll
    for (int mt = 0; mt < 4; ++mt)
#pragma unroll
        for (int ntl = 0; ntl < NTW; ++ntl) acc[mt][ntl] = vzero;

#pragma unroll
    for (int ntl = 0; ntl < NTW; ++ntl) {
        const int nt = wave * NTW + ntl;
        const unsigned short* bbase = Bt + (size_t)(nt * 16 + bcol) * K + bkh;
#pragma unroll 1
        for (int kt = 0; kt < KT; ++kt) {
            const int kb = kt * 32;
            v16us braw = *(const v16us*)(bbase + kb);   // 2 x global b128
            v16bf b = __builtin_bit_cast(v16bf, braw);
#pragma unroll
            for (int mt = 0; mt < 4; ++mt) {
                const unsigned short* ap = &As[(mt * 16 + arow) * LDA + kb + akh];
                v8us a0 = *(const v8us*)ap;             // ds b128
                v8us a1 = *(const v8us*)(ap + 8);       // ds b128
                v16bf a = __builtin_bit_cast(
                    v16bf, __builtin_shufflevector(a0, a1, 0, 1, 2, 3, 4, 5, 6, 7,
                                                   8, 9, 10, 11, 12, 13, 14, 15));
                acc[mt][ntl] = __builtin_amdgcn_wmma_f32_16x16x32_bf16(
                    false, a, false, b, (short)0, acc[mt][ntl], false, false);
            }
        }
    }

    // ---- store C ----
    const int cn = lane & 15;
    const int cm = (lane >> 4) * 8;
#pragma unroll
    for (int ntl = 0; ntl < NTW; ++ntl) {
        const int n0 = (wave * NTW + ntl) * 16 + cn;
#pragma unroll
        for (int mt = 0; mt < 4; ++mt) {
#pragma unroll
            for (int i = 0; i < 8; ++i) {
                int row = m00 + mt * 16 + cm + i;
                if (row < M) C[(size_t)row * NCOLS + n0] = acc[mt][ntl][i];
            }
        }
    }
}

// ---------------------------------------------------------------------------
// 4) SpMM (gather by dst row) + bias + ReLU, store bf16 for GEMM2
// ---------------------------------------------------------------------------
__global__ __launch_bounds__(256)
void spmm_relu_bf16_kernel(const float* __restrict__ XW1,
                           const int* __restrict__ off,
                           const int* __restrict__ eidx,
                           const int* __restrict__ esrc,
                           const float* __restrict__ ew,
                           const float* __restrict__ b1,
                           unsigned short* __restrict__ hb) {
    const int row = blockIdx.x;
    const int f = threadIdx.x;  // 0..255
    __shared__ int   s_src[128];
    __shared__ float s_w[128];
    const int beg = off[row], end = off[row + 1];
    float acc = 0.f;
    for (int base = beg; base < end; base += 128) {
        const int cnt = min(128, end - base);
        if (f < cnt) {
            const int e = eidx[base + f];
            s_src[f] = esrc[e];
            s_w[f]   = ew[e];
        }
        __syncthreads();
        for (int j = 0; j < cnt; ++j)
            acc = fmaf(s_w[j], XW1[(size_t)s_src[j] * 256 + f], acc);
        __syncthreads();
    }
    float h = acc + b1[f];
    h = h > 0.f ? h : 0.f;
    hb[(size_t)row * 256 + f] = f32_to_bf16_rne(h);
}

// ---------------------------------------------------------------------------
// 5) SpMM for logits + bias + fused log_softmax over 64 classes
// ---------------------------------------------------------------------------
__global__ __launch_bounds__(64)
void spmm_logsoftmax_kernel(const float* __restrict__ HW2,
                            const int* __restrict__ off,
                            const int* __restrict__ eidx,
                            const int* __restrict__ esrc,
                            const float* __restrict__ ew,
                            const float* __restrict__ b2,
                            float* __restrict__ out) {
    const int row = blockIdx.x;
    const int f = threadIdx.x;  // 0..63
    __shared__ int   s_src[64];
    __shared__ float s_w[64];
    __shared__ float red[64];
    const int beg = off[row], end = off[row + 1];
    float acc = 0.f;
    for (int base = beg; base < end; base += 64) {
        const int cnt = min(64, end - base);
        if (f < cnt) {
            const int e = eidx[base + f];
            s_src[f] = esrc[e];
            s_w[f]   = ew[e];
        }
        __syncthreads();
        for (int j = 0; j < cnt; ++j)
            acc = fmaf(s_w[j], HW2[(size_t)s_src[j] * 64 + f], acc);
        __syncthreads();
    }
    const float logit = acc + b2[f];

    red[f] = logit;
    __syncthreads();
#pragma unroll
    for (int s = 32; s > 0; s >>= 1) {
        if (f < s) red[f] = fmaxf(red[f], red[f + s]);
        __syncthreads();
    }
    const float mx = red[0];
    __syncthreads();
    red[f] = expf(logit - mx);
    __syncthreads();
#pragma unroll
    for (int s = 32; s > 0; s >>= 1) {
        if (f < s) red[f] += red[f + s];
        __syncthreads();
    }
    const float lse = mx + logf(red[0]);
    out[(size_t)row * 64 + f] = logit - lse;
}

// ---------------------------------------------------------------------------
// Host launcher
// ---------------------------------------------------------------------------
extern "C" void kernel_launch(void* const* d_in, const int* in_sizes, int n_in,
                              void* d_out, int out_size, void* d_ws, size_t ws_size,
                              hipStream_t stream) {
    const float* x    = (const float*)d_in[0];
    const int*   esrc = (const int*)d_in[1];
    const int*   edst = (const int*)d_in[2];
    const float* ew   = (const float*)d_in[3];
    const float* W1   = (const float*)d_in[4];
    const float* b1   = (const float*)d_in[5];
    const float* W2   = (const float*)d_in[6];
    const float* b2   = (const float*)d_in[7];
    float* out = (float*)d_out;

    const int NFEAT = 512, NHID = 256, NCLASS = 64;
    const int N = in_sizes[0] / NFEAT;
    const int E = in_sizes[1];

    // workspace layout (256B aligned slabs)
    char* ws = (char*)d_ws;
    size_t o = 0;
    auto take = [&](size_t bytes) -> char* {
        char* p = ws + o;
        o += (bytes + 255) & ~(size_t)255;
        return p;
    };
    int* deg    = (int*)take((size_t)N * 4);
    int* off    = (int*)take((size_t)(N + 1) * 4);
    int* cursor = (int*)take((size_t)N * 4);
    int* eidx   = (int*)take((size_t)E * 4);
    unsigned short* W1t = (unsigned short*)take((size_t)NFEAT * NHID * 2);   // [NHID][NFEAT]
    unsigned short* W2t = (unsigned short*)take((size_t)NHID * NCLASS * 2);  // [NCLASS][NHID]
    float* XW1  = (float*)take((size_t)N * NHID * 4);
    unsigned short* hb = (unsigned short*)take((size_t)N * NHID * 2);
    float* HW2  = (float*)take((size_t)N * NCLASS * 4);
    (void)ws_size; (void)n_in; (void)out_size;

    // 1) weights -> bf16, transposed to [N][K] for contiguous B fragments
    transpose_bf16_kernel<<<(NFEAT * NHID + 255) / 256, 256, 0, stream>>>(
        W1, W1t, NFEAT, NHID);
    transpose_bf16_kernel<<<(NHID * NCLASS + 255) / 256, 256, 0, stream>>>(
        W2, W2t, NHID, NCLASS);

    // 2) CSR build by destination
    zero_int_kernel<<<(N + 255) / 256, 256, 0, stream>>>(deg, N);
    hist_kernel<<<(E + 255) / 256, 256, 0, stream>>>(edst, deg, E);
    scan_kernel<<<1, 1024, 0, stream>>>(deg, off, cursor, N);
    scatter_kernel<<<(E + 255) / 256, 256, 0, stream>>>(edst, cursor, eidx, E);

    const int mblocks = (N + 63) / 64;
    // 3) XW1 = x @ W1   (fp32 A staged->bf16 in LDS, bf16 WMMA, fp32 acc)
    wmma_gemm_tiled<512, 256, true>
        <<<mblocks, 128, 0, stream>>>(x, W1t, XW1, N);
    // 4) h = relu(spmm(XW1) + b1), stored bf16
    spmm_relu_bf16_kernel<<<N, 256, 0, stream>>>(XW1, off, eidx, esrc, ew, b1, hb);
    // 5) HW2 = h @ W2
    wmma_gemm_tiled<256, 64, false>
        <<<mblocks, 128, 0, stream>>>(hb, W2t, HW2, N);
    // 6) logits = spmm(HW2) + b2 ; out = log_softmax(logits)
    spmm_logsoftmax_kernel<<<N, 64, 0, stream>>>(HW2, off, eidx, esrc, ew, b2, out);
}
